// UVSTModule_5128190951866
// MI455X (gfx1250) — matrix operand order repaired and checked by reading, not compiled, not run on hardware
//
#include <hip/hip_runtime.h>
#include <hip/hip_bf16.h>

// ---------------------------------------------------------------------------
// USVT via Gram-matrix eigensolve:
//   M = (2x-1)*mask  [224x672]   G = M M^T  [224x224]
//   G = U diag(lam) U^T (Jacobi), keep top-179 eigvecs -> w mask
//   T = (diag(w) U^T) M ;  Wrec = U T ;  out = (clip(Wrec/p_obs)+1)/2
// GEMMs run on v_wmma_f32_16x16x32_f16 with hi/lo f16 error compensation.
// ---------------------------------------------------------------------------

typedef _Float16 h16;
typedef __attribute__((ext_vector_type(16))) _Float16 v16h;
typedef __attribute__((ext_vector_type(8)))  _Float16 v8h;
typedef __attribute__((ext_vector_type(8)))  float    v8f;

#define BATCH  64
#define HH     224          // rows of M (= H)
#define CW     672          // cols of M (= C*W)
#define NM     (HH*CW)      // 150528 elems of M per batch
#define NG     (HH*HH)      // 50176 elems of G/U per batch
#define KTRUNC 179          // int(0.8*224)
#define SWEEPS 8
#define NPAIR  112          // HH/2 disjoint Jacobi pairs per round

// ---- WMMA fragment loaders (16-bit A 16x32 / B 32x16 ISA lane layouts) ----

__device__ __forceinline__ v16h load_a16(const h16* __restrict__ base, int stride,
                                         int m0, int k0, int lane) {
  // A (MxK=16x32): lanes 0-15 row m0+lane hold K {0..7,16..23};
  //                lanes 16-31 row m0+lane-16 hold K {8..15,24..31}
  int row = m0 + (lane & 15);
  int kb  = k0 + ((lane & 16) ? 8 : 0);
  const h16* p = base + (size_t)row * stride + kb;
  v8h x0 = *(const v8h*)(p);
  v8h x1 = *(const v8h*)(p + 16);
  return __builtin_shufflevector(x0, x1, 0,1,2,3,4,5,6,7,8,9,10,11,12,13,14,15);
}

__device__ __forceinline__ v16h load_b16(const h16* __restrict__ baseT, int stride,
                                         int n0, int k0, int lane) {
  // B (KxN=32x16), column n striped per lane: lanes 0-15 hold K {0..7,16..23} of col
  // n0+lane; lanes 16-31 hold K {8..15,24..31} of col n0+lane-16. baseT is stored so
  // that column n of B is a contiguous row (we pre-transpose operands).
  int row = n0 + (lane & 15);
  int kb  = k0 + ((lane & 16) ? 8 : 0);
  const h16* p = baseT + (size_t)row * stride + kb;
  v8h x0 = *(const v8h*)(p);
  v8h x1 = *(const v8h*)(p + 16);
  return __builtin_shufflevector(x0, x1, 0,1,2,3,4,5,6,7,8,9,10,11,12,13,14,15);
}

__device__ __forceinline__ v8f wmma3(v16h ah, v16h al, v16h bh, v16h bl, v8f acc) {
  // split-compensated f32-accuracy product: hi*hi + hi*lo + lo*hi
  acc = __builtin_amdgcn_wmma_f32_16x16x32_f16(false, ah, false, bh, (short)0, acc, false, false);
  acc = __builtin_amdgcn_wmma_f32_16x16x32_f16(false, ah, false, bl, (short)0, acc, false, false);
  acc = __builtin_amdgcn_wmma_f32_16x16x32_f16(false, al, false, bh, (short)0, acc, false, false);
  return acc;
}

// ---------------------------------------------------------------------------
// Kernel 0: zero per-batch mask-count accumulators
__global__ void k_zero(float* __restrict__ p_sum) {
  if (threadIdx.x < BATCH) p_sum[threadIdx.x] = 0.f;
}

// Kernel 1: build masked matrix M in hi/lo f16 (row-major + transposed) + p_obs sums
__global__ void k_prep(const float* __restrict__ x, const int* __restrict__ mask,
                       h16* __restrict__ Mh, h16* __restrict__ Ml,
                       h16* __restrict__ Mth, h16* __restrict__ Mtl,
                       float* __restrict__ p_sum) {
  int e  = blockIdx.x * 256 + threadIdx.x;     // < BATCH*NM ; blocks never straddle batches
  int b  = e / NM, r = e % NM;
  int h  = r / CW, cw = r % CW;
  int c  = cw / 224, w_ = cw % 224;
  int m  = mask[e];                             // mask flat layout == [B,H,CW]
  float v = 0.f;
  if (m) v = 2.f * x[(((size_t)b * 3 + c) * HH + h) * 224 + w_] - 1.f;
  h16 hi = (h16)v;
  h16 lo = (h16)(v - (float)hi);
  Mh[e] = hi;  Ml[e] = lo;
  size_t t = (size_t)b * NM + (size_t)cw * HH + h;   // M^T [672x224]
  Mth[t] = hi; Mtl[t] = lo;

  __shared__ float red[256];
  red[threadIdx.x] = (float)m;
  __syncthreads();
  for (int s = 128; s > 0; s >>= 1) {
    if (threadIdx.x < s) red[threadIdx.x] += red[threadIdx.x + s];
    __syncthreads();
  }
  if (threadIdx.x == 0) atomicAdd(&p_sum[b], red[0]);
}

// Kernel 2: G = M M^T  (WMMA, 14x14 tiles of 16x16, K=672)
__global__ void k_gram(const h16* __restrict__ Mh, const h16* __restrict__ Ml,
                       float* __restrict__ G) {
  int b    = blockIdx.y;
  int tile = blockIdx.x * 4 + (threadIdx.x >> 5);   // 0..195 (4 waves/block)
  int lane = threadIdx.x & 31;
  int m0 = (tile / 14) * 16, n0 = (tile % 14) * 16;
  const h16* Ah = Mh + (size_t)b * NM;
  const h16* Al = Ml + (size_t)b * NM;
  v8f acc = {};
  for (int k0 = 0; k0 < CW; k0 += 32) {
    if (k0 + 64 <= CW) {
      __builtin_prefetch(Ah + (size_t)(m0 + (lane & 15)) * CW + k0 + 64, 0, 1);
      __builtin_prefetch(Ah + (size_t)(n0 + (lane & 15)) * CW + k0 + 64, 0, 1);
    }
    v16h ah = load_a16(Ah, CW, m0, k0, lane);
    v16h al = load_a16(Al, CW, m0, k0, lane);
    v16h bh = load_b16(Ah, CW, n0, k0, lane);   // B[k][n] = M[n][k]: rows of M
    v16h bl = load_b16(Al, CW, n0, k0, lane);
    acc = wmma3(ah, al, bh, bl, acc);
  }
  float* Gb = G + (size_t)b * NG;
  int n  = n0 + (lane & 15);
  int mb = m0 + ((lane & 16) ? 8 : 0);
  for (int j = 0; j < 8; ++j) Gb[(size_t)(mb + j) * HH + n] = acc[j];
}

// Kernel 3: two-sided cyclic Jacobi eigensolver. G lives in 200KB dynamic LDS
// (CDNA5: 320KB/WGP); eigenvector accumulator U stays global (L2-resident).
__global__ __launch_bounds__(1024) void k_jacobi(float* __restrict__ G,
                                                 float* __restrict__ U) {
  extern __shared__ float Gs[];                 // HH*HH floats = 200704 B
  __shared__ float cA[NPAIR], sA[NPAIR];
  __shared__ int   pA[NPAIR], qA[NPAIR];
  int b   = blockIdx.x;
  int tid = threadIdx.x;
  float* Gb = G + (size_t)b * NG;
  float* Ub = U + (size_t)b * NG;

  for (int i = tid; i < NG; i += 1024) {
    Gs[i] = Gb[i];
    Ub[i] = ((i / HH) == (i % HH)) ? 1.f : 0.f;
  }
  __syncthreads();

  for (int sweep = 0; sweep < SWEEPS; ++sweep) {
    for (int r = 0; r < HH - 1; ++r) {
      if (tid < NPAIR) {                        // round-robin disjoint pairing
        int p, q;
        if (tid == 0) { p = HH - 1; q = r % (HH - 1); }
        else { p = (r + tid) % (HH - 1); q = (r + (HH - 1) - tid) % (HH - 1); }
        float app = Gs[p * HH + p], aqq = Gs[q * HH + q], apq = Gs[p * HH + q];
        float c = 1.f, s = 0.f;
        if (fabsf(apq) > 1e-12f) {
          float tau = (aqq - app) / (2.f * apq);
          float t   = 1.f / (fabsf(tau) + sqrtf(1.f + tau * tau));
          if (tau < 0.f) t = -t;
          c = rsqrtf(1.f + t * t);
          s = t * c;
        }
        cA[tid] = c; sA[tid] = s; pA[tid] = p; qA[tid] = q;
      }
      __syncthreads();
      for (int i = tid; i < NPAIR * HH; i += 1024) {      // row pass: J^T G
        int pr = i / HH, col = i % HH;
        float c = cA[pr], s = sA[pr]; int p = pA[pr], q = qA[pr];
        float gp = Gs[p * HH + col], gq = Gs[q * HH + col];
        Gs[p * HH + col] = c * gp - s * gq;
        Gs[q * HH + col] = s * gp + c * gq;
      }
      __syncthreads();
      for (int i = tid; i < NPAIR * HH; i += 1024) {      // col pass: (.)J  +  U <- U J
        int pr = i / HH, row = i % HH;
        float c = cA[pr], s = sA[pr]; int p = pA[pr], q = qA[pr];
        float gp = Gs[row * HH + p], gq = Gs[row * HH + q];
        Gs[row * HH + p] = c * gp - s * gq;
        Gs[row * HH + q] = s * gp + c * gq;
        float up = Ub[row * HH + p], uq = Ub[row * HH + q];
        Ub[row * HH + p] = c * up - s * uq;
        Ub[row * HH + q] = s * up + c * uq;
      }
      __syncthreads();
    }
  }
  if (tid < HH) Gb[tid * HH + tid] = Gs[tid * HH + tid];  // eigenvalues out
}

// Kernel 4: top-k selection (rank counting) + convert U and w-masked U^T to hi/lo f16
__global__ void k_select(const float* __restrict__ G, const float* __restrict__ U,
                         h16* __restrict__ Uh, h16* __restrict__ Ul,
                         h16* __restrict__ UTh, h16* __restrict__ UTl) {
  int b = blockIdx.x, tid = threadIdx.x;                  // 256 threads
  const float* Gb = G + (size_t)b * NG;
  const float* Ub = U + (size_t)b * NG;
  __shared__ float d[HH];
  __shared__ float w[HH];
  if (tid < HH) d[tid] = Gb[tid * HH + tid];
  __syncthreads();
  if (tid < HH) {
    float di = d[tid];
    int rank = 0;
    for (int j = 0; j < HH; ++j) {
      float dj = d[j];
      rank += (dj > di) || (dj == di && j < tid);
    }
    w[tid] = (rank < KTRUNC) ? 1.f : 0.f;
  }
  __syncthreads();
  for (int i = tid; i < NG; i += 256) {
    float u  = Ub[i];
    h16 uhi  = (h16)u;
    Uh[(size_t)b * NG + i] = uhi;
    Ul[(size_t)b * NG + i] = (h16)(u - (float)uhi);
    int rrow = i / HH, hcol = i % HH;                     // (r,h) for U^T
    float ut = w[rrow] * Ub[hcol * HH + rrow];
    h16 thi  = (h16)ut;
    UTh[(size_t)b * NG + i] = thi;
    UTl[(size_t)b * NG + i] = (h16)(ut - (float)thi);
  }
}

// Kernel 5: T = (diag(w) U^T) M, stored transposed (Tt [672x224]) in hi/lo f16
__global__ void k_proj1(const h16* __restrict__ UTh, const h16* __restrict__ UTl,
                        const h16* __restrict__ Mth, const h16* __restrict__ Mtl,
                        h16* __restrict__ Tth, h16* __restrict__ Ttl) {
  int b    = blockIdx.y;
  int tile = blockIdx.x * 4 + (threadIdx.x >> 5);         // 0..587
  int lane = threadIdx.x & 31;
  int m0 = (tile / 42) * 16, n0 = (tile % 42) * 16;       // m: 224 rows of T, n: 672
  const h16* Ahp = UTh + (size_t)b * NG;
  const h16* Alp = UTl + (size_t)b * NG;
  const h16* Bhp = Mth + (size_t)b * NM;                  // B[k][n]=M[k][n]=Mt[n][k]
  const h16* Blp = Mtl + (size_t)b * NM;
  v8f acc = {};
  for (int k0 = 0; k0 < HH; k0 += 32) {
    if (k0 + 64 <= HH) __builtin_prefetch(Bhp + (size_t)(n0 + (lane & 15)) * HH + k0 + 64, 0, 1);
    v16h ah = load_a16(Ahp, HH, m0, k0, lane);
    v16h al = load_a16(Alp, HH, m0, k0, lane);
    v16h bh = load_b16(Bhp, HH, n0, k0, lane);
    v16h bl = load_b16(Blp, HH, n0, k0, lane);
    acc = wmma3(ah, al, bh, bl, acc);
  }
  int n  = n0 + (lane & 15);
  int mb = m0 + ((lane & 16) ? 8 : 0);
  h16* TH = Tth + (size_t)b * NM;
  h16* TL = Ttl + (size_t)b * NM;
  for (int j = 0; j < 8; ++j) {                           // store transposed Tt[n][m]
    float v = acc[j];
    h16 hi = (h16)v;
    TH[(size_t)n * HH + (mb + j)] = hi;
    TL[(size_t)n * HH + (mb + j)] = (h16)(v - (float)hi);
  }
}

// Kernel 6: Wrec = U T, fused epilogue: /p_obs, clip, (v+1)/2, NCHW scatter
__global__ void k_proj2(const h16* __restrict__ Uh, const h16* __restrict__ Ul,
                        const h16* __restrict__ Tth, const h16* __restrict__ Ttl,
                        const float* __restrict__ p_sum, float* __restrict__ out) {
  int b    = blockIdx.y;
  int tile = blockIdx.x * 4 + (threadIdx.x >> 5);         // 0..587
  int lane = threadIdx.x & 31;
  int m0 = (tile / 42) * 16, n0 = (tile % 42) * 16;       // m: h (224), n: cw (672)
  const h16* Ahp = Uh  + (size_t)b * NG;
  const h16* Alp = Ul  + (size_t)b * NG;
  const h16* Bhp = Tth + (size_t)b * NM;                  // B[k][n]=T[k][n]=Tt[n][k]
  const h16* Blp = Ttl + (size_t)b * NM;
  v8f acc = {};
  for (int k0 = 0; k0 < HH; k0 += 32) {
    if (k0 + 64 <= HH) __builtin_prefetch(Bhp + (size_t)(n0 + (lane & 15)) * HH + k0 + 64, 0, 1);
    v16h ah = load_a16(Ahp, HH, m0, k0, lane);
    v16h al = load_a16(Alp, HH, m0, k0, lane);
    v16h bh = load_b16(Bhp, HH, n0, k0, lane);
    v16h bl = load_b16(Blp, HH, n0, k0, lane);
    acc = wmma3(ah, al, bh, bl, acc);
  }
  float p   = p_sum[b] * (1.f / (float)NM);
  float inv = 1.f / fmaxf(p, 1e-8f);
  int n  = n0 + (lane & 15);
  int mb = m0 + ((lane & 16) ? 8 : 0);
  int c  = n / 224, w_ = n % 224;
  for (int j = 0; j < 8; ++j) {
    float v = acc[j] * inv;
    v = fminf(1.f, fmaxf(-1.f, v));
    int h = mb + j;
    out[(((size_t)b * 3 + c) * HH + h) * 224 + w_] = (v + 1.f) * 0.5f;
  }
}

// ---------------------------------------------------------------------------
extern "C" void kernel_launch(void* const* d_in, const int* in_sizes, int n_in,
                              void* d_out, int out_size, void* d_ws, size_t ws_size,
                              hipStream_t stream) {
  (void)in_sizes; (void)n_in; (void)out_size; (void)ws_size;
  const float* x   = (const float*)d_in[0];
  const int* mask  = (const int*)d_in[1];
  float* out       = (float*)d_out;

  char* ws = (char*)d_ws;
  size_t off = 0;
  auto carve = [&](size_t bytes) -> void* {
    void* p = ws + off;
    off = (off + bytes + 255) & ~(size_t)255;
    return p;
  };
  h16*   Mh    = (h16*)carve((size_t)BATCH * NM * sizeof(h16));
  h16*   Ml    = (h16*)carve((size_t)BATCH * NM * sizeof(h16));
  h16*   Mth   = (h16*)carve((size_t)BATCH * NM * sizeof(h16));
  h16*   Mtl   = (h16*)carve((size_t)BATCH * NM * sizeof(h16));
  float* G     = (float*)carve((size_t)BATCH * NG * sizeof(float));
  float* U     = (float*)carve((size_t)BATCH * NG * sizeof(float));
  h16*   Uh    = (h16*)carve((size_t)BATCH * NG * sizeof(h16));
  h16*   Ul    = (h16*)carve((size_t)BATCH * NG * sizeof(h16));
  h16*   UTh   = (h16*)carve((size_t)BATCH * NG * sizeof(h16));
  h16*   UTl   = (h16*)carve((size_t)BATCH * NG * sizeof(h16));
  h16*   Tth   = (h16*)carve((size_t)BATCH * NM * sizeof(h16));
  h16*   Ttl   = (h16*)carve((size_t)BATCH * NM * sizeof(h16));
  float* p_sum = (float*)carve(BATCH * sizeof(float));

  k_zero  <<<1, 64, 0, stream>>>(p_sum);
  k_prep  <<<(BATCH * NM) / 256, 256, 0, stream>>>(x, mask, Mh, Ml, Mth, Mtl, p_sum);
  k_gram  <<<dim3(49, BATCH), 128, 0, stream>>>(Mh, Ml, G);
  k_jacobi<<<BATCH, 1024, HH * HH * sizeof(float), stream>>>(G, U);
  k_select<<<BATCH, 256, 0, stream>>>(G, U, Uh, Ul, UTh, UTl);
  k_proj1 <<<dim3(147, BATCH), 128, 0, stream>>>(UTh, UTl, Mth, Mtl, Tth, Ttl);
  k_proj2 <<<dim3(147, BATCH), 128, 0, stream>>>(Uh, Ul, Tth, Ttl, p_sum, out);
}
